// BitConv2d_22393959482132
// MI455X (gfx1250) — matrix operand order repaired and checked by reading, not compiled, not run on hardware
//
#include <hip/hip_runtime.h>

// BitConv2d for MI455X (gfx1250): 8 bit-plane 3x3 convs as one implicit GEMM.
// x/weight tiles staged once feed 8 v_wmma_f32_16x16x32_f16 (one per plane).
// A elements are bit * 2^-14 (one shl + one and per packed pair), weights
// pre-scaled by 2^10 (exact), so acc == conv(bit_plane / 16) as in the
// reference. Weight staging uses GLOBAL_LOAD_ASYNC_TO_LDS_B128 (ASYNCcnt).

typedef __attribute__((ext_vector_type(16))) _Float16 v16h;
typedef __attribute__((ext_vector_type(8)))  unsigned v8u;
typedef __attribute__((ext_vector_type(8)))  float    v8f;

#define BATCH 16
#define CIN   256
#define HH    56
#define WWID  56
#define HWSZ  (HH * WWID)      // 3136
#define COUT  256
#define TAPS  9
#define BITS  8

#define PIX_TILE 32            // pixels per workgroup (2 M-tiles of 16)
#define CO_TILE  64            // output channels per workgroup (4 N-tiles)
#define CI_BLK   32            // K-chunk of input channels
#define NCIBLK   (CIN / CI_BLK)
#define XPIX     146           // staged pixels: hw0-57 .. hw0+31+57
#define XPST     20            // u32 per pixel row (16 pairs + pad, 16B rows)
#define WCHUNKS  (CO_TILE * TAPS * CI_BLK * 2 / 16)   // 2304 x 16B

// ---- one-shot prep: w -> f16(w * 2^10), relayout [blk][co][tap][ci'] ----
__global__ __launch_bounds__(256) void weight_prep_kernel(
    const float* __restrict__ w, _Float16* __restrict__ wt) {
  int idx = blockIdx.x * 256 + threadIdx.x;        // linear over [co][ci][r]
  if (idx >= COUT * CIN * TAPS) return;
  int co  = idx / (CIN * TAPS);
  int rem = idx - co * (CIN * TAPS);
  int ci  = rem / TAPS;
  int r   = rem - ci * TAPS;
  int blk = ci >> 5, cl = ci & 31;
  wt[(((size_t)blk * COUT + co) * TAPS + r) * CI_BLK + cl] =
      (_Float16)(w[idx] * 1024.0f);                // *2^10 exact in f16 range
}

__global__ __launch_bounds__(256) void bitconv_wmma_kernel(
    const float* __restrict__ x, const _Float16* __restrict__ wh,
    const float* __restrict__ bias, float* __restrict__ y)
{
  __shared__ __align__(16) unsigned xp[XPIX][XPST];            // 11,680 B
  __shared__ __align__(16) _Float16 ws[CO_TILE][TAPS][CI_BLK]; // 36,864 B

  const int t      = threadIdx.x;
  const int lane   = t & 31;          // wave32
  const int wave   = t >> 5;          // 8 waves
  const int wave_m = wave & 1;
  const int wave_n = wave >> 1;

  const int hw0 = blockIdx.x * PIX_TILE;   // 98
  const int co0 = blockIdx.y * CO_TILE;    // 4
  const int b   = blockIdx.z;              // 16

  const int mrow     = lane & 15;
  const int m_pix    = wave_m * 16 + mrow;
  const int hw       = hw0 + m_pix;
  const int ph       = hw / WWID;
  const int pw       = hw % WWID;
  const int kg2      = (lane >> 4) * 8;    // A K-group base in u32-pairs
  const int kb       = (lane >> 4) * 16;   // B K base (contiguous, k = 16h + s)
  const int co_local = wave_n * 16 + mrow;

  v8f acc[BITS];
  #pragma unroll
  for (int k = 0; k < BITS; ++k) acc[k] = (v8f){0.f,0.f,0.f,0.f,0.f,0.f,0.f,0.f};

  const float* xb = x + (size_t)b * CIN * HWSZ;
  const unsigned lds_ws = (unsigned)(uintptr_t)&ws[0][0][0];

  for (int blk = 0; blk < NCIBLK; ++blk) {
    const int ci0 = blk * CI_BLK;
    __syncthreads();

    // ---- stage weights: async DMA global -> LDS, no VGPR round-trip ----
    {
      const void* src = (const void*)(wh + ((size_t)blk * COUT + co0) *
                                           (TAPS * CI_BLK));
      #pragma unroll
      for (int j = t; j < WCHUNKS; j += 256) {
        unsigned voff = (unsigned)j * 16u;
        unsigned ldsa = lds_ws + voff;
        asm volatile("global_load_async_to_lds_b128 %0, %1, %2"
                     :: "v"(ldsa), "v"(voff), "s"(src) : "memory");
      }
    }
    // ---- stage x: f32 -> int, two channels packed per u32 ----
    {
      const int cp  = t >> 4;                      // channel-pair 0..15
      const int l16 = t & 15;
      const float* xc = xb + (size_t)(ci0 + 2 * cp) * HWSZ;
      for (int p = l16; p < XPIX; p += 16) {
        int gp = hw0 - 57 + p;                     // clamped; borders masked later
        gp = gp < 0 ? 0 : (gp >= HWSZ ? HWSZ - 1 : gp);
        int a0 = (int)xc[gp];
        int a1 = (int)xc[gp + HWSZ];
        xp[p][cp] = (unsigned)(a0 | (a1 << 16));
      }
    }
    asm volatile("s_wait_asynccnt 0x0" ::: "memory");
    __syncthreads();

    #pragma unroll
    for (int r = 0; r < TAPS; ++r) {
      const int dh = r / 3 - 1, dw = r % 3 - 1;
      const bool valid = ((unsigned)(ph + dh) < (unsigned)HH) &&
                         ((unsigned)(pw + dw) < (unsigned)WWID);
      const unsigned mval = valid ? 0x04000400u : 0u;   // mask folds validity
      const int poff = m_pix + 57 + dh * WWID + dw;

      // A tile: 2 x ds_load_b128 per lane
      const uint4 qa = *(const uint4*)&xp[poff][kg2];
      const uint4 qb = *(const uint4*)&xp[poff][4 + kg2];
      const unsigned q[8] = {qa.x, qa.y, qa.z, qa.w, qb.x, qb.y, qb.z, qb.w};

      // B tile: lanes 0-15 K=0..15, lanes 16-31 K=16..31; N = lane&15
      v16h bmat = *(const v16h*)&ws[co_local][r][kb];

      #pragma unroll
      for (int k = 0; k < BITS; ++k) {
        const int shl = 3 + k;                    // bit (7-k) -> f16 bit 10
        v8u au;
        #pragma unroll
        for (int j = 0; j < 8; ++j)
          au[j] = (q[j] << shl) & mval;           // bit -> f16 2^-14 per half
        v16h amat = __builtin_bit_cast(v16h, au);
        acc[k] = __builtin_amdgcn_wmma_f32_16x16x32_f16(
            false, amat, false, bmat, (short)0, acc[k], false, false);
      }
    }

    // hint next x block toward the caches (global_prefetch_b8)
    if (blk + 1 < NCIBLK) {
      const int ci = t >> 3;
      __builtin_prefetch(xb + (size_t)(ci0 + CI_BLK + ci) * HWSZ + hw0, 0, 3);
    }
  }

  // ---- epilogue: per-plane trunc, shift-accumulate, bias, LDS transpose ----
  __syncthreads();
  float* myt = (float*)&xp[0][0] + wave * (16 * 17);   // per-wave 16x17 slice
  const float bval = bias[co0 + co_local];
  #pragma unroll
  for (int rr = 0; rr < 8; ++rr) {
    const int m = rr + (lane >> 4) * 8;          // C/D row per doc layout
    float v = 0.0f;
    #pragma unroll
    for (int k = 0; k < BITS; ++k)
      v += truncf(acc[k][rr]) * (float)(1 << (7 - k));
    v = v * 16.0f + bval;                         // * N_SCALE + bias
    myt[mrow * 17 + m] = v;
  }
  __syncthreads();

  // coalesced stores: lanes 0-15 sweep contiguous hw for a fixed cout
  float* yb = y + ((size_t)b * COUT + co0 + wave_n * 16) * HWSZ
                + hw0 + wave_m * 16;
  #pragma unroll
  for (int i = 0; i < 8; ++i) {
    const int cl = i * 2 + (lane >> 4);
    const int m  = lane & 15;
    yb[(size_t)cl * HWSZ + m] = myt[cl * 17 + m];
  }
}

extern "C" void kernel_launch(void* const* d_in, const int* in_sizes, int n_in,
                              void* d_out, int out_size, void* d_ws, size_t ws_size,
                              hipStream_t stream) {
  (void)in_sizes; (void)n_in; (void)ws_size; (void)out_size;
  const float* x    = (const float*)d_in[0];   // [16,256,56,56]
  const float* wgt  = (const float*)d_in[1];   // [256,256,3,3]
  const float* bias = (const float*)d_in[2];   // [1,256,1,1]
  float* y = (float*)d_out;                    // [16,256,56,56]
  _Float16* wt = (_Float16*)d_ws;              // 1.18 MB f16 weights (L2-hot)

  int nw = COUT * CIN * TAPS;                  // 589,824
  weight_prep_kernel<<<(nw + 255) / 256, 256, 0, stream>>>(wgt, wt);

  dim3 grid(HWSZ / PIX_TILE, COUT / CO_TILE, BATCH);  // (98, 4, 16)
  bitconv_wmma_kernel<<<grid, 256, 0, stream>>>(x, wt, bias, y);
}
